// Seq2SeqBahAtt_31181462569445
// MI455X (gfx1250) — matrix operand order repaired and checked by reading, not compiled, not run on hardware
//
#include <hip/hip_runtime.h>
#include <hip/hip_bf16.h>

// ---------------- model dims ----------------
#define Bc     16
#define Sc     128
#define CMAXc  16
#define WEc    256
#define Hc     256
#define Ac     512
#define H2c    512
#define NCLSc  2000

typedef __attribute__((ext_vector_type(16))) __bf16        bf16x16;
typedef __attribute__((ext_vector_type(8)))  float         f32x8;
typedef __attribute__((ext_vector_type(8)))  unsigned int  u32x8;

__device__ __forceinline__ unsigned bf16bits(float f) {
  unsigned u = __builtin_bit_cast(unsigned, f);
  u += 0x7FFFu + ((u >> 16) & 1u);           // round-to-nearest-even
  return u >> 16;
}

__device__ __forceinline__ __bf16 f2bf(float f) {
  unsigned short s = (unsigned short)bf16bits(f);
  return __builtin_bit_cast(__bf16, s);
}

__device__ __forceinline__ float sigmoidf(float x) {
  return 1.0f / (1.0f + __expf(-x));
}

// ---------------------------------------------------------------------------
// Pack W[N,K] (f32, row-major) into bf16 WMMA B-fragments.
// Fragment element e of (ntile, kt, lane): K = kt*32 + (lane>>4)*16 + e,
//                                          N = ntile*16 + (lane&15).
// Stored at P[((ntile*(K/32) + kt)*32 + lane)*16 + e]; rows n >= Nreal -> 0.
// grid.x = (Npad/16)*(K/32), block = 32
// ---------------------------------------------------------------------------
__global__ __launch_bounds__(32) void pack_wT_bf16(
    const float* __restrict__ W, int Nreal, int K,
    unsigned short* __restrict__ P)
{
  const int lane = threadIdx.x;
  const int half = lane >> 4, l16 = lane & 15;
  const int nkt  = K >> 5;
  const int ntile = blockIdx.x / nkt, kt = blockIdx.x % nkt;
  const int n = ntile * 16 + l16;

  unsigned int v[8];
  if (n < Nreal) {
    const float* src = W + (size_t)n * K + kt * 32 + half * 16;
#pragma unroll
    for (int i = 0; i < 8; ++i) {
      const unsigned lo = bf16bits(src[2 * i]);
      const unsigned hi = bf16bits(src[2 * i + 1]);
      v[i] = lo | (hi << 16);
    }
  } else {
#pragma unroll
    for (int i = 0; i < 8; ++i) v[i] = 0u;
  }
  uint4* dst = (uint4*)(P + ((size_t)blockIdx.x * 32 + lane) * 16);
  dst[0] = make_uint4(v[0], v[1], v[2], v[3]);
  dst[1] = make_uint4(v[4], v[5], v[6], v[7]);
}

// ---------------------------------------------------------------------------
// WMMA GEMM with pre-packed bf16 weights:
//   C[M, 0..Nreal) = A[M,K] * W^T (+ bias) (+= C if accum)
// One wave computes a 16x64 strip (4 N-tiles, 4 independent accumulators).
// grid = (Npad/64, M/16), block = 32.
// A fragment per lane = one contiguous 16-float run -> 4x b128 loads + cvt.
// B fragment per lane = 16 contiguous bf16 -> 2x b128 loads, no cvt.
// ---------------------------------------------------------------------------
__global__ __launch_bounds__(32) void gemm_pk(
    const float* __restrict__ A, int lda,
    const unsigned short* __restrict__ P,
    const float* __restrict__ bias,
    float* __restrict__ C, int ldc,
    int K, int Nreal, int accum)
{
  const int lane = threadIdx.x;
  const int half = lane >> 4, l16 = lane & 15;
  const int nkt  = K >> 5;

  const float* __restrict__ Ar =
      A + (size_t)(blockIdx.y * 16 + l16) * lda + half * 16;
  const unsigned short* __restrict__ Pb =
      P + ((size_t)blockIdx.x * 4 * nkt * 32 + lane) * 16;

  const f32x8 zero = {0.f, 0.f, 0.f, 0.f, 0.f, 0.f, 0.f, 0.f};
  f32x8 acc[4];
#pragma unroll
  for (int j = 0; j < 4; ++j) acc[j] = zero;

  for (int kt = 0; kt < nkt; ++kt) {
    const float4 r0 = *(const float4*)(Ar + kt * 32);
    const float4 r1 = *(const float4*)(Ar + kt * 32 + 4);
    const float4 r2 = *(const float4*)(Ar + kt * 32 + 8);
    const float4 r3 = *(const float4*)(Ar + kt * 32 + 12);
    if (kt + 1 < nkt) __builtin_prefetch(Ar + (kt + 1) * 32, 0, 3);

    const float fa[16] = {r0.x, r0.y, r0.z, r0.w, r1.x, r1.y, r1.z, r1.w,
                          r2.x, r2.y, r2.z, r2.w, r3.x, r3.y, r3.z, r3.w};
    bf16x16 af;
#pragma unroll
    for (int e = 0; e < 16; ++e) af[e] = f2bf(fa[e]);

#pragma unroll
    for (int j = 0; j < 4; ++j) {
      const unsigned short* p = Pb + (size_t)(j * nkt + kt) * 32 * 16;
      const uint4 b0 = *(const uint4*)p;
      const uint4 b1 = *(const uint4*)(p + 8);
      const u32x8 u = {b0.x, b0.y, b0.z, b0.w, b1.x, b1.y, b1.z, b1.w};
      const bf16x16 bfr = __builtin_bit_cast(bf16x16, u);
      acc[j] = __builtin_amdgcn_wmma_f32_16x16x32_bf16(
          false, af, false, bfr, (short)0, acc[j], false, false);
    }
  }

#pragma unroll
  for (int j = 0; j < 4; ++j) {
    const int n = (blockIdx.x * 4 + j) * 16 + l16;
    if (n < Nreal) {
      const float bv = bias ? bias[n] : 0.0f;
#pragma unroll
      for (int r = 0; r < 8; ++r) {   // VGPR r: M = half*8 + r, N = l16
        float* cp = C + (size_t)(blockIdx.y * 16 + half * 8 + r) * ldc + n;
        float v = acc[j][r] + bv;
        if (accum) v += *cp;
        *cp = v;
      }
    }
  }
}

// ---------------------------------------------------------------------------
// Embedding gather: out[r,:] = emb[idx[r*idx_stride+idx_off], :]
// ---------------------------------------------------------------------------
__global__ void gather_emb(const float* __restrict__ emb,
                           const int* __restrict__ idx,
                           int idx_stride, int idx_off,
                           float* __restrict__ out, int dim)
{
  const int r = blockIdx.x, j = threadIdx.x;
  const int id = idx[r * idx_stride + idx_off];
  out[(long)r * dim + j] = emb[(long)id * dim + j];
}

// decoder-input embedding: xc[b,0:WE] = dec_emb[t==0 ? 0 : targets[b,t-1]]
__global__ void dec_gather(const float* __restrict__ emb,
                           const int* __restrict__ targets, int t,
                           float* __restrict__ xc, int xcstride)
{
  const int b = blockIdx.x, j = threadIdx.x;
  const int id = (t == 0) ? 0 : targets[b * Sc + t - 1];
  xc[(long)b * xcstride + j] = emb[(long)id * WEc + j];
}

// reverse within length: dst[b,t,:] = src[b, clip(len-1-t,0,T-1), :] (* mask)
__global__ void reverse_rows(const float* __restrict__ src,
                             const int* __restrict__ lens,
                             float* __restrict__ dst, int T, int dim,
                             int apply_mask)
{
  const int b = blockIdx.x / T, t = blockIdx.x % T, j = threadIdx.x;
  const int len = lens[b];
  int idx = len - 1 - t;
  idx = idx < 0 ? 0 : (idx > T - 1 ? T - 1 : idx);
  float v = src[((long)b * T + idx) * dim + j];
  if (apply_mask && t >= len) v = 0.0f;
  dst[((long)b * T + t) * dim + j] = v;
}

// masked LSTM gate update (torch order i,f,g,o). g0 row = gates for this step.
__global__ void lstm_elem(const float* __restrict__ g0, long gstride,
                          const int* __restrict__ lens, int t,
                          float* __restrict__ h, float* __restrict__ c,
                          float* __restrict__ y0, long ystride,
                          int N, int Hh)
{
  const int idx = blockIdx.x * blockDim.x + threadIdx.x;
  if (idx >= N * Hh) return;
  const int n = idx / Hh, j = idx % Hh;
  const bool m = t < lens[n];
  const float* g = g0 + (long)n * gstride;
  const float ig = sigmoidf(g[j]);
  const float fg = sigmoidf(g[Hh + j]);
  const float gg = tanhf(g[2 * Hh + j]);
  const float og = sigmoidf(g[3 * Hh + j]);
  const float cn = fg * c[idx] + ig * gg;
  const float hn = og * tanhf(cn);
  if (m) { c[idx] = cn; h[idx] = hn; }
  if (y0) y0[(long)n * ystride + j] = m ? hn : 0.0f;
}

// enc[r,:] = concat(yf[r,:], yb[r,:]) + chT[r,:]   (r = b*S+s)
__global__ void enc_combine(const float* __restrict__ yf,
                            const float* __restrict__ yb,
                            const float* __restrict__ chT,
                            float* __restrict__ enc)
{
  const int r = blockIdx.x, j = threadIdx.x;
  const float v = (j < Hc) ? yf[(long)r * Hc + j] : yb[(long)r * Hc + j - Hc];
  enc[(long)r * H2c + j] = v + chT[(long)r * H2c + j];
}

__global__ void copy_h0(const float* __restrict__ enc, float* __restrict__ h)
{
  const int b = blockIdx.x, j = threadIdx.x;
  h[(long)b * H2c + j] = enc[((long)b * Sc + Sc - 1) * H2c + j];
}

// e[b,s] = scale * dot(k[b,s,:], q[b,:]) + (s<len ? 0 : NEG); 1 wave per (b,s)
__global__ __launch_bounds__(32) void attn_scores(
    const float* __restrict__ k, const float* __restrict__ q,
    const int* __restrict__ words_num, float* __restrict__ e)
{
  const int bs = blockIdx.x, b = bs / Sc, s = bs % Sc, lane = threadIdx.x;
  const float* kk = k + (long)bs * Ac;
  const float* qq = q + (long)b * Ac;
  float sum = 0.0f;
  for (int j = lane; j < Ac; j += 32) sum += kk[j] * qq[j];
  for (int off = 16; off; off >>= 1) sum += __shfl_down(sum, off, 32);
  if (lane == 0) {
    const float scale = 0.04419417382415922f;  // 1/sqrt(512)
    const float neg = (s < words_num[b]) ? 0.0f : -1.0e9f;
    e[bs] = sum * scale + neg;
  }
}

__global__ void softmax_rows(const float* __restrict__ e, float* __restrict__ a)
{
  __shared__ float sm[Sc];
  const int b = blockIdx.x, s = threadIdx.x;
  const float v = e[b * Sc + s];
  sm[s] = v; __syncthreads();
  for (int off = Sc / 2; off; off >>= 1) {
    if (s < off) sm[s] = fmaxf(sm[s], sm[s + off]);
    __syncthreads();
  }
  const float mx = sm[0]; __syncthreads();
  const float ex = __expf(v - mx);
  sm[s] = ex; __syncthreads();
  for (int off = Sc / 2; off; off >>= 1) {
    if (s < off) sm[s] += sm[s + off];
    __syncthreads();
  }
  a[b * Sc + s] = ex / sm[0];
}

// xc[b, WE + a] = sum_s alpha[b,s] * v[b,s,a]
__global__ void attn_context(const float* __restrict__ alpha,
                             const float* __restrict__ v,
                             float* __restrict__ xc, int xcstride)
{
  const int idx = blockIdx.x * blockDim.x + threadIdx.x;
  if (idx >= Bc * Ac) return;
  const int b = idx / Ac, a = idx % Ac;
  float s = 0.0f;
  for (int t = 0; t < Sc; ++t)
    s += alpha[b * Sc + t] * v[((long)b * Sc + t) * Ac + a];
  xc[(long)b * xcstride + WEc + a] = s;
}

__global__ void gru_elem(const float* __restrict__ gi,
                         const float* __restrict__ gh,
                         float* __restrict__ h)
{
  const int idx = blockIdx.x * blockDim.x + threadIdx.x;
  if (idx >= Bc * H2c) return;
  const int b = idx / H2c, j = idx % H2c;
  const float* gib = gi + (long)b * 3 * H2c;
  const float* ghb = gh + (long)b * 3 * H2c;
  const float r = sigmoidf(gib[j] + ghb[j]);
  const float z = sigmoidf(gib[H2c + j] + ghb[H2c + j]);
  const float n = tanhf(gib[2 * H2c + j] + r * ghb[2 * H2c + j]);
  h[idx] = (1.0f - z) * n + z * h[idx];
}

// ---------------------------------------------------------------------------
extern "C" void kernel_launch(void* const* d_in, const int* in_sizes, int n_in,
                              void* d_out, int out_size, void* d_ws, size_t ws_size,
                              hipStream_t stream)
{
  (void)out_size; (void)ws_size; (void)n_in;
  auto F  = [&](int i) { return (const float*)d_in[i]; };
  auto Ii = [&](int i) { return (const int*)d_in[i]; };

  const float *word_emb, *char_emb, *wl_Wih_f, *wl_Whh_f, *wl_b_f,
      *wl_Wih_b, *wl_Whh_b, *wl_b_b, *cl_Wih, *cl_Whh, *cl_b,
      *Wq, *bq, *Wk, *bk, *Wv, *bv, *dec_emb,
      *gru_Wih, *gru_Whh, *gru_bih, *gru_bhh, *Wout, *bout;
  const int *words, *words_num, *chars, *char_lens, *targets;

  if (in_sizes[0] == 12800000) {
    // params dict in insertion order, then the remaining inputs
    word_emb = F(0);  char_emb = F(1);  wl_Wih_f = F(2);  wl_Whh_f = F(3);
    wl_b_f = F(4);    wl_Wih_b = F(5);  wl_Whh_b = F(6);  wl_b_b = F(7);
    cl_Wih = F(8);    cl_Whh = F(9);    cl_b = F(10);     Wq = F(11);
    bq = F(12);       Wk = F(13);       bk = F(14);       Wv = F(15);
    bv = F(16);       dec_emb = F(17);  gru_Wih = F(18);  gru_Whh = F(19);
    gru_bih = F(20);  gru_bhh = F(21);  Wout = F(22);     bout = F(23);
    words = Ii(24); words_num = Ii(25); chars = Ii(26); char_lens = Ii(27);
    /* mask = 28 (ignored) */ targets = Ii(29);
  } else if (in_sizes[0] == 2048 && in_sizes[1] == 32768) {
    // jax tree_leaves: everything sorted by key
    char_lens = Ii(0); chars = Ii(1); /* mask = 2 (ignored) */
    Wk = F(3);  Wout = F(4);  Wq = F(5);  Wv = F(6);
    bk = F(7);  bout = F(8);  bq = F(9);  bv = F(10);
    char_emb = F(11); cl_Whh = F(12); cl_Wih = F(13); cl_b = F(14);
    dec_emb = F(15);  gru_Whh = F(16); gru_Wih = F(17);
    gru_bhh = F(18);  gru_bih = F(19);
    wl_Whh_b = F(20); wl_Whh_f = F(21); wl_Wih_b = F(22); wl_Wih_f = F(23);
    wl_b_b = F(24);   wl_b_f = F(25);   word_emb = F(26);
    targets = Ii(27); words = Ii(28); words_num = Ii(29);
  } else {
    // params sorted by key, remaining inputs in insertion order
    Wk = F(0);  Wout = F(1);  Wq = F(2);  Wv = F(3);
    bk = F(4);  bout = F(5);  bq = F(6);  bv = F(7);
    char_emb = F(8); cl_Whh = F(9); cl_Wih = F(10); cl_b = F(11);
    dec_emb = F(12); gru_Whh = F(13); gru_Wih = F(14);
    gru_bhh = F(15); gru_bih = F(16);
    wl_Whh_b = F(17); wl_Whh_f = F(18); wl_Wih_b = F(19); wl_Wih_f = F(20);
    wl_b_b = F(21);   wl_b_f = F(22);   word_emb = F(23);
    words = Ii(24); words_num = Ii(25); chars = Ii(26); char_lens = Ii(27);
    /* mask = 28 */ targets = Ii(29);
  }

  float* out = (float*)d_out;  // [B, S, NCLS] fp32

  // ---------------- workspace layout ----------------
  size_t off = 0;  // in floats
  auto alloc = [&](size_t n) { float* p = (float*)d_ws + off; off += n; return p; };
  float* x      = alloc((size_t)Bc * Sc * WEc);
  float* x_rev  = alloc((size_t)Bc * Sc * WEc);
  float* gi_f   = alloc((size_t)Bc * Sc * 4 * Hc);
  float* gi_b   = alloc((size_t)Bc * Sc * 4 * Hc);
  float* yf     = alloc((size_t)Bc * Sc * Hc);
  float* yb_rev = alloc((size_t)Bc * Sc * Hc);
  float* yb     = alloc((size_t)Bc * Sc * Hc);
  float* h_f    = alloc((size_t)Bc * Hc);
  float* c_f    = alloc((size_t)Bc * Hc);
  float* h_b    = alloc((size_t)Bc * Hc);
  float* c_b    = alloc((size_t)Bc * Hc);
  float* ce_t   = alloc((size_t)Bc * Sc * WEc);           // [2048, 256]
  float* cgi    = alloc((size_t)Bc * Sc * 4 * H2c);       // [2048, 2048]
  float* ch     = alloc((size_t)Bc * Sc * H2c);           // [2048, 512]
  float* cc     = alloc((size_t)Bc * Sc * H2c);
  float* enc    = alloc((size_t)Bc * Sc * H2c);
  float* kbuf   = alloc((size_t)Bc * Sc * Ac);
  float* vbuf   = alloc((size_t)Bc * Sc * Ac);
  float* hdec   = alloc((size_t)Bc * H2c);
  float* qbuf   = alloc((size_t)Bc * Ac);
  float* ebuf   = alloc((size_t)Bc * Sc);
  float* abuf   = alloc((size_t)Bc * Sc);
  float* xc     = alloc((size_t)Bc * (WEc + Ac));         // [16, 768]
  float* dgi    = alloc((size_t)Bc * 3 * H2c);
  float* dgh    = alloc((size_t)Bc * 3 * H2c);

  // packed bf16 weights (ushort regions carved from the same float arena)
  auto allocU = [&](size_t nu) {
    unsigned short* p = (unsigned short*)((float*)d_ws + off);
    off += nu / 2;  // nu is always even
    return p;
  };
  auto pack = [&](const float* W, int Nreal, int K, int Npad) {
    unsigned short* P = allocU((size_t)Npad * K);
    pack_wT_bf16<<<(Npad / 16) * (K / 32), 32, 0, stream>>>(W, Nreal, K, P);
    return P;
  };
  unsigned short* P_wih_f = pack(wl_Wih_f, 4 * Hc, WEc, 4 * Hc);
  unsigned short* P_whh_f = pack(wl_Whh_f, 4 * Hc, Hc,  4 * Hc);
  unsigned short* P_wih_b = pack(wl_Wih_b, 4 * Hc, WEc, 4 * Hc);
  unsigned short* P_whh_b = pack(wl_Whh_b, 4 * Hc, Hc,  4 * Hc);
  unsigned short* P_cih   = pack(cl_Wih,  4 * H2c, WEc, 4 * H2c);
  unsigned short* P_chh   = pack(cl_Whh,  4 * H2c, H2c, 4 * H2c);
  unsigned short* P_q     = pack(Wq, Ac, H2c, Ac);
  unsigned short* P_k     = pack(Wk, Ac, H2c, Ac);
  unsigned short* P_v     = pack(Wv, Ac, H2c, Ac);
  unsigned short* P_gih   = pack(gru_Wih, 3 * H2c, WEc + Ac, 3 * H2c);
  unsigned short* P_ghh   = pack(gru_Whh, 3 * H2c, H2c,      3 * H2c);
  unsigned short* P_out   = pack(Wout, NCLSc, H2c, 2048);   // pad 2000 -> 2048

  auto gemm = [&](const float* A, int lda, const unsigned short* P,
                  const float* bias, float* C, int ldc,
                  int M, int Nreal, int Npad, int K, int accum) {
    dim3 g(Npad / 64, M / 16);
    gemm_pk<<<g, 32, 0, stream>>>(A, lda, P, bias, C, ldc, K, Nreal, accum);
  };

  // =================== encoder: word BiLSTM ===================
  gather_emb<<<Bc * Sc, WEc, 0, stream>>>(word_emb, words, 1, 0, x, WEc);
  reverse_rows<<<Bc * Sc, WEc, 0, stream>>>(x, words_num, x_rev, Sc, WEc, 0);

  // batched input projections for all timesteps
  gemm(x,     WEc, P_wih_f, wl_b_f, gi_f, 4 * Hc, Bc * Sc, 4 * Hc, 4 * Hc, WEc, 0);
  gemm(x_rev, WEc, P_wih_b, wl_b_b, gi_b, 4 * Hc, Bc * Sc, 4 * Hc, 4 * Hc, WEc, 0);

  hipMemsetAsync(h_f, 0, (size_t)Bc * Hc * 4, stream);
  hipMemsetAsync(c_f, 0, (size_t)Bc * Hc * 4, stream);
  hipMemsetAsync(h_b, 0, (size_t)Bc * Hc * 4, stream);
  hipMemsetAsync(c_b, 0, (size_t)Bc * Hc * 4, stream);

  const long wg_stride = (long)Sc * 4 * Hc;   // gate-row stride per batch row
  const long wy_stride = (long)Sc * Hc;
  for (int t = 0; t < Sc; ++t) {
    gemm(h_f, Hc, P_whh_f, nullptr, gi_f + (long)t * 4 * Hc, (int)wg_stride,
         Bc, 4 * Hc, 4 * Hc, Hc, 1);
    lstm_elem<<<(Bc * Hc + 255) / 256, 256, 0, stream>>>(
        gi_f + (long)t * 4 * Hc, wg_stride, words_num, t, h_f, c_f,
        yf + (long)t * Hc, wy_stride, Bc, Hc);
    gemm(h_b, Hc, P_whh_b, nullptr, gi_b + (long)t * 4 * Hc, (int)wg_stride,
         Bc, 4 * Hc, 4 * Hc, Hc, 1);
    lstm_elem<<<(Bc * Hc + 255) / 256, 256, 0, stream>>>(
        gi_b + (long)t * 4 * Hc, wg_stride, words_num, t, h_b, c_b,
        yb_rev + (long)t * Hc, wy_stride, Bc, Hc);
  }
  reverse_rows<<<Bc * Sc, Hc, 0, stream>>>(yb_rev, words_num, yb, Sc, Hc, 1);

  // =================== encoder: char LSTM ===================
  hipMemsetAsync(ch, 0, (size_t)Bc * Sc * H2c * 4, stream);
  hipMemsetAsync(cc, 0, (size_t)Bc * Sc * H2c * 4, stream);
  for (int t = 0; t < CMAXc; ++t) {
    gather_emb<<<Bc * Sc, WEc, 0, stream>>>(char_emb, chars, CMAXc, t, ce_t, WEc);
    gemm(ce_t, WEc, P_cih, cl_b,  cgi, 4 * H2c, Bc * Sc, 4 * H2c, 4 * H2c, WEc, 0);
    gemm(ch,   H2c, P_chh, nullptr, cgi, 4 * H2c, Bc * Sc, 4 * H2c, 4 * H2c, H2c, 1);
    lstm_elem<<<(Bc * Sc * H2c + 255) / 256, 256, 0, stream>>>(
        cgi, (long)4 * H2c, char_lens, t, ch, cc, nullptr, 0, Bc * Sc, H2c);
  }

  // =================== enc + attention K/V ===================
  enc_combine<<<Bc * Sc, H2c, 0, stream>>>(yf, yb, ch, enc);
  gemm(enc, H2c, P_k, bk, kbuf, Ac, Bc * Sc, Ac, Ac, H2c, 0);
  gemm(enc, H2c, P_v, bv, vbuf, Ac, Bc * Sc, Ac, Ac, H2c, 0);
  copy_h0<<<Bc, H2c, 0, stream>>>(enc, hdec);

  // =================== decoder ===================
  for (int t = 0; t < Sc; ++t) {
    dec_gather<<<Bc, WEc, 0, stream>>>(dec_emb, targets, t, xc, WEc + Ac);
    gemm(hdec, H2c, P_q, bq, qbuf, Ac, Bc, Ac, Ac, H2c, 0);
    attn_scores<<<Bc * Sc, 32, 0, stream>>>(kbuf, qbuf, words_num, ebuf);
    softmax_rows<<<Bc, Sc, 0, stream>>>(ebuf, abuf);
    attn_context<<<(Bc * Ac + 255) / 256, 256, 0, stream>>>(abuf, vbuf, xc, WEc + Ac);
    gemm(xc,   WEc + Ac, P_gih, gru_bih, dgi, 3 * H2c, Bc, 3 * H2c, 3 * H2c, WEc + Ac, 0);
    gemm(hdec, H2c,      P_ghh, gru_bhh, dgh, 3 * H2c, Bc, 3 * H2c, 3 * H2c, H2c, 0);
    gru_elem<<<(Bc * H2c + 255) / 256, 256, 0, stream>>>(dgi, dgh, hdec);
    // out[:, t, :] = hdec @ Wout^T + bout
    gemm(hdec, H2c, P_out, bout, out + (long)t * NCLSc, Sc * NCLSc,
         Bc, NCLSc, 2048, H2c, 0);
  }
}
